// Encoder_76390288327304
// MI455X (gfx1250) — matrix operand order repaired and checked by reading, not compiled, not run on hardware
//
#include <hip/hip_runtime.h>
#include <hip/hip_bf16.h>

typedef _Float16 half_t;
typedef __attribute__((ext_vector_type(16))) _Float16 v16h;
typedef __attribute__((ext_vector_type(8)))  _Float16 v8h;
typedef __attribute__((ext_vector_type(4)))  _Float16 v4h;
typedef __attribute__((ext_vector_type(8)))  float    v8f;

#define DIM   128
#define E_TOT 500000
#define NGRID 200000
#define NMESH 40000
#define LN_EPS 1e-5f

// ---------------- fragment loaders (CDNA5 WMMA 16x16x32 f16 layouts) --------

// A: 16x32 f16. lane<16: row=lane, VGPR0-3 K=0..7, VGPR4-7 K=16..23
//               lane>=16: row=lane-16, VGPR0-3 K=8..15, VGPR4-7 K=24..31
__device__ __forceinline__ v16h load_a_frag(const half_t* __restrict__ base,
                                            int stride, int lane, int kbase) {
    const int hf  = (lane >> 4) & 1;
    const int row = lane & 15;
    const half_t* p = base + row * stride + kbase + hf * 8;
    v8h lo = *(const v8h*)(p);
    v8h hi = *(const v8h*)(p + 16);
    v16h a;
#pragma unroll
    for (int i = 0; i < 8; ++i) { a[i] = lo[i]; a[8 + i] = hi[i]; }
    return a;
}

// B: 32x16 f16 from weights pre-transposed to [N][K]. lanes 0-15: col=lane,
// K=0..7,16..23; lanes 16-31: col=lane-16, K=8..15,24..31. Two 16B loads.
__device__ __forceinline__ v16h load_b_frag(const half_t* __restrict__ wt,
                                            int K, int lane, int col0, int kbase) {
    const int n  = lane & 15;
    const int kh = (lane >> 4) & 1;
    const half_t* p = wt + (size_t)(col0 + n) * K + kbase + kh * 8;
    v8h lo = *(const v8h*)(p);
    v8h hi = *(const v8h*)(p + 16);
    v16h b;
#pragma unroll
    for (int i = 0; i < 8; ++i) { b[i] = lo[i]; b[8 + i] = hi[i]; }
    return b;
}

__device__ __forceinline__ float wave_sum(float x) {
#pragma unroll
    for (int m = 16; m >= 1; m >>= 1) x += __shfl_xor(x, m, 32);
    return x;
}

// Two-row-tile GEMM slab: acc{0,1} += A[{0,16}..][K] * B[K][col0..col0+16)
// K-outer with dual accumulators so each B fragment is loaded once and feeds
// two back-to-back WMMAs.
template <int KSTEPS>
__device__ __forceinline__ void gemm_slab(const half_t* __restrict__ s_a, int astride,
                                          const half_t* __restrict__ wt, int K,
                                          int lane, int col0,
                                          v8f& acc0, v8f& acc1) {
#pragma unroll
    for (int ks = 0; ks < KSTEPS; ++ks) {
        v16h b  = load_b_frag(wt, K, lane, col0, ks * 32);
        v16h a0 = load_a_frag(s_a,                astride, lane, ks * 32);
        v16h a1 = load_a_frag(s_a + 16 * astride, astride, lane, ks * 32);
        acc0 = __builtin_amdgcn_wmma_f32_16x16x32_f16(false, a0, false, b,
                                                      (short)0, acc0, false, false);
        acc1 = __builtin_amdgcn_wmma_f32_16x16x32_f16(false, a1, false, b,
                                                      (short)0, acc1, false, false);
    }
}

// ---------------- prep kernels ----------------------------------------------

__global__ __launch_bounds__(256) void transpose_f16(
        const float* __restrict__ w, half_t* __restrict__ wt,
        int K, int N, int total) {
    int idx = blockIdx.x * blockDim.x + threadIdx.x;
    if (idx < total) {
        int k = idx / N;
        int n = idx - k * N;
        wt[(size_t)n * K + k] = (half_t)w[idx];
    }
}

__global__ __launch_bounds__(256) void zero_f32(float* __restrict__ p, int total) {
    int idx = blockIdx.x * blockDim.x + threadIdx.x;
    if (idx < total) p[idx] = 0.0f;
}

// ---------------- edge MLP + scatter-add ------------------------------------
// 32 edges per block; 8 waves, each owns 16 output columns.

__global__ __launch_bounds__(256) void edge_mlp_scatter(
    const float* __restrict__ efeat, const float* __restrict__ gfeat,
    const float* __restrict__ mfeat, const int* __restrict__ src,
    const int* __restrict__ dst,
    const half_t* __restrict__ w1t, const float* __restrict__ b1,
    const half_t* __restrict__ w2t, const float* __restrict__ b2,
    const float* __restrict__ gam, const float* __restrict__ bet,
    float* __restrict__ agg)
{
    const int TILE = 32;
    const int CSTR = 392;   // 384 + 8 pad (16B-aligned rows, spreads banks)
    const int HSTR = 136;
    const int YSTR = 132;
    __shared__ int    s_src[TILE], s_dst[TILE];
    __shared__ half_t s_cat[TILE * CSTR];
    __shared__ half_t s_h[TILE * HSTR];
    __shared__ float  s_y[TILE * YSTR];

    const int tid  = threadIdx.x;
    const int lane = tid & 31;
    const int wv   = tid >> 5;
    const int e0   = blockIdx.x * TILE;

    if (tid < TILE) {
        s_src[tid] = src[e0 + tid];
        s_dst[tid] = dst[e0 + tid];
    }
    __syncthreads();

    // stage concat [32 x 384] as f16; 96 float4 per row, 3072 total
#pragma unroll
    for (int i = 0; i < 12; ++i) {
        int idx = tid + i * 256;
        int e = idx / 96;
        int q = idx - e * 96;
        const float4* sp;
        if (q < 32)      sp = (const float4*)(efeat + (size_t)(e0 + e) * DIM) + q;
        else if (q < 64) sp = (const float4*)(gfeat + (size_t)s_src[e] * DIM) + (q - 32);
        else             sp = (const float4*)(mfeat + (size_t)s_dst[e] * DIM) + (q - 64);
        float4 v = *sp;
        v4h hv; hv[0] = (half_t)v.x; hv[1] = (half_t)v.y;
        hv[2] = (half_t)v.z; hv[3] = (half_t)v.w;
        *(v4h*)(s_cat + e * CSTR + q * 4) = hv;
    }
    __syncthreads();

    const int col0 = wv * 16;
    const int n    = lane & 15;
    const int hf   = (lane >> 4) & 1;

    // layer 1: h = silu(cat @ w1 + b1), K = 384
    {
        const float bias1 = b1[col0 + n];
        v8f acc0 = {}, acc1 = {};
        gemm_slab<12>(s_cat, CSTR, w1t, 384, lane, col0, acc0, acc1);
#pragma unroll
        for (int r = 0; r < 8; ++r) {
            float y0 = acc0[r] + bias1;
            float y1 = acc1[r] + bias1;
            float h0 = y0 / (1.0f + __expf(-y0));   // silu
            float h1 = y1 / (1.0f + __expf(-y1));
            s_h[(r + 8 * hf) * HSTR + col0 + n]        = (half_t)h0;
            s_h[(16 + r + 8 * hf) * HSTR + col0 + n]   = (half_t)h1;
        }
    }
    __syncthreads();

    // layer 2: y = h @ w2 + b2, K = 128
    {
        const float bias2 = b2[col0 + n];
        v8f acc0 = {}, acc1 = {};
        gemm_slab<4>(s_h, HSTR, w2t, 128, lane, col0, acc0, acc1);
#pragma unroll
        for (int r = 0; r < 8; ++r) {
            s_y[(r + 8 * hf) * YSTR + col0 + n]      = acc0[r] + bias2;
            s_y[(16 + r + 8 * hf) * YSTR + col0 + n] = acc1[r] + bias2;
        }
    }
    __syncthreads();

    // layernorm per row + atomic scatter-add to agg[dst]; 4 rows per wave
#pragma unroll
    for (int rr = 0; rr < 4; ++rr) {
        int row = wv * 4 + rr;
        float v0 = s_y[row * YSTR + lane];
        float v1 = s_y[row * YSTR + lane + 32];
        float v2 = s_y[row * YSTR + lane + 64];
        float v3 = s_y[row * YSTR + lane + 96];
        float s  = wave_sum(v0 + v1 + v2 + v3);
        float sq = wave_sum(v0 * v0 + v1 * v1 + v2 * v2 + v3 * v3);
        float mu  = s * (1.0f / 128.0f);
        float var = sq * (1.0f / 128.0f) - mu * mu;
        float rs  = rsqrtf(var + LN_EPS);
        float* dp = agg + (size_t)s_dst[row] * DIM;
        float vv[4] = {v0, v1, v2, v3};
#pragma unroll
        for (int j = 0; j < 4; ++j) {
            int c = lane + 32 * j;
            float val = (vv[j] - mu) * rs * gam[c] + bet[c];
            atomicAdd(dp + c, val);
        }
    }
}

// ---------------- grid node MLP (K=128) with residual -----------------------

__global__ __launch_bounds__(256) void grid_mlp(
    const float* __restrict__ x,
    const half_t* __restrict__ w1t, const float* __restrict__ b1,
    const half_t* __restrict__ w2t, const float* __restrict__ b2,
    const float* __restrict__ gam, const float* __restrict__ bet,
    float* __restrict__ out)
{
    const int TILE = 32, XSTR = 136, HSTR = 136, YSTR = 132;
    __shared__ half_t s_x[TILE * XSTR];
    __shared__ half_t s_h[TILE * HSTR];
    __shared__ float  s_y[TILE * YSTR];

    const int tid  = threadIdx.x;
    const int lane = tid & 31;
    const int wv   = tid >> 5;
    const size_t r0 = (size_t)blockIdx.x * TILE;

#pragma unroll
    for (int i = 0; i < 4; ++i) {
        int idx = tid + i * 256;
        int e = idx >> 5;
        int q = idx & 31;
        float4 v = *((const float4*)(x + (r0 + e) * DIM) + q);
        v4h hv; hv[0] = (half_t)v.x; hv[1] = (half_t)v.y;
        hv[2] = (half_t)v.z; hv[3] = (half_t)v.w;
        *(v4h*)(s_x + e * XSTR + q * 4) = hv;
    }
    __syncthreads();

    const int col0 = wv * 16;
    const int n    = lane & 15;
    const int hf   = (lane >> 4) & 1;

    {
        const float bias1 = b1[col0 + n];
        v8f acc0 = {}, acc1 = {};
        gemm_slab<4>(s_x, XSTR, w1t, 128, lane, col0, acc0, acc1);
#pragma unroll
        for (int r = 0; r < 8; ++r) {
            float y0 = acc0[r] + bias1;
            float y1 = acc1[r] + bias1;
            float h0 = y0 / (1.0f + __expf(-y0));
            float h1 = y1 / (1.0f + __expf(-y1));
            s_h[(r + 8 * hf) * HSTR + col0 + n]      = (half_t)h0;
            s_h[(16 + r + 8 * hf) * HSTR + col0 + n] = (half_t)h1;
        }
    }
    __syncthreads();

    {
        const float bias2 = b2[col0 + n];
        v8f acc0 = {}, acc1 = {};
        gemm_slab<4>(s_h, HSTR, w2t, 128, lane, col0, acc0, acc1);
#pragma unroll
        for (int r = 0; r < 8; ++r) {
            s_y[(r + 8 * hf) * YSTR + col0 + n]      = acc0[r] + bias2;
            s_y[(16 + r + 8 * hf) * YSTR + col0 + n] = acc1[r] + bias2;
        }
    }
    __syncthreads();

#pragma unroll
    for (int rr = 0; rr < 4; ++rr) {
        int row = wv * 4 + rr;
        float v0 = s_y[row * YSTR + lane];
        float v1 = s_y[row * YSTR + lane + 32];
        float v2 = s_y[row * YSTR + lane + 64];
        float v3 = s_y[row * YSTR + lane + 96];
        float s  = wave_sum(v0 + v1 + v2 + v3);
        float sq = wave_sum(v0 * v0 + v1 * v1 + v2 * v2 + v3 * v3);
        float mu  = s * (1.0f / 128.0f);
        float var = sq * (1.0f / 128.0f) - mu * mu;
        float rs  = rsqrtf(var + LN_EPS);
        float vv[4] = {v0, v1, v2, v3};
        size_t rb = (r0 + row) * DIM;
#pragma unroll
        for (int j = 0; j < 4; ++j) {
            int c = lane + 32 * j;
            float val = (vv[j] - mu) * rs * gam[c] + bet[c];
            out[rb + c] = x[rb + c] + val;   // residual
        }
    }
}

// ---------------- mesh node MLP (K=256: [agg || mesh]) with residual --------
// `agg` aliases `out` (mesh half of d_out): read during staging, write at end.

__global__ __launch_bounds__(256) void mesh_mlp(
    const float* __restrict__ agg, const float* __restrict__ mfeat,
    const half_t* __restrict__ w1t, const float* __restrict__ b1,
    const half_t* __restrict__ w2t, const float* __restrict__ b2,
    const float* __restrict__ gam, const float* __restrict__ bet,
    float* __restrict__ out)
{
    const int TILE = 32, CSTR = 264, HSTR = 136, YSTR = 132;
    __shared__ half_t s_cat[TILE * CSTR];
    __shared__ half_t s_h[TILE * HSTR];
    __shared__ float  s_y[TILE * YSTR];

    const int tid  = threadIdx.x;
    const int lane = tid & 31;
    const int wv   = tid >> 5;
    const size_t r0 = (size_t)blockIdx.x * TILE;

    // stage [agg || mesh] : 64 float4 per row, 2048 total
#pragma unroll
    for (int i = 0; i < 8; ++i) {
        int idx = tid + i * 256;
        int e = idx >> 6;
        int q = idx & 63;
        const float4* sp;
        if (q < 32) sp = (const float4*)(agg   + (r0 + e) * DIM) + q;
        else        sp = (const float4*)(mfeat + (r0 + e) * DIM) + (q - 32);
        float4 v = *sp;
        v4h hv; hv[0] = (half_t)v.x; hv[1] = (half_t)v.y;
        hv[2] = (half_t)v.z; hv[3] = (half_t)v.w;
        *(v4h*)(s_cat + e * CSTR + q * 4) = hv;
    }
    __syncthreads();

    const int col0 = wv * 16;
    const int n    = lane & 15;
    const int hf   = (lane >> 4) & 1;

    {
        const float bias1 = b1[col0 + n];
        v8f acc0 = {}, acc1 = {};
        gemm_slab<8>(s_cat, CSTR, w1t, 256, lane, col0, acc0, acc1);
#pragma unroll
        for (int r = 0; r < 8; ++r) {
            float y0 = acc0[r] + bias1;
            float y1 = acc1[r] + bias1;
            float h0 = y0 / (1.0f + __expf(-y0));
            float h1 = y1 / (1.0f + __expf(-y1));
            s_h[(r + 8 * hf) * HSTR + col0 + n]      = (half_t)h0;
            s_h[(16 + r + 8 * hf) * HSTR + col0 + n] = (half_t)h1;
        }
    }
    __syncthreads();

    {
        const float bias2 = b2[col0 + n];
        v8f acc0 = {}, acc1 = {};
        gemm_slab<4>(s_h, HSTR, w2t, 128, lane, col0, acc0, acc1);
#pragma unroll
        for (int r = 0; r < 8; ++r) {
            s_y[(r + 8 * hf) * YSTR + col0 + n]      = acc0[r] + bias2;
            s_y[(16 + r + 8 * hf) * YSTR + col0 + n] = acc1[r] + bias2;
        }
    }
    __syncthreads();

#pragma unroll
    for (int rr = 0; rr < 4; ++rr) {
        int row = wv * 4 + rr;
        float v0 = s_y[row * YSTR + lane];
        float v1 = s_y[row * YSTR + lane + 32];
        float v2 = s_y[row * YSTR + lane + 64];
        float v3 = s_y[row * YSTR + lane + 96];
        float s  = wave_sum(v0 + v1 + v2 + v3);
        float sq = wave_sum(v0 * v0 + v1 * v1 + v2 * v2 + v3 * v3);
        float mu  = s * (1.0f / 128.0f);
        float var = sq * (1.0f / 128.0f) - mu * mu;
        float rs  = rsqrtf(var + LN_EPS);
        float vv[4] = {v0, v1, v2, v3};
        size_t rb = (r0 + row) * DIM;
#pragma unroll
        for (int j = 0; j < 4; ++j) {
            int c = lane + 32 * j;
            float val = (vv[j] - mu) * rs * gam[c] + bet[c];
            out[rb + c] = mfeat[rb + c] + val;   // residual (overwrites agg)
        }
    }
}

// ---------------- launch ----------------------------------------------------

extern "C" void kernel_launch(void* const* d_in, const int* in_sizes, int n_in,
                              void* d_out, int out_size, void* d_ws, size_t ws_size,
                              hipStream_t stream) {
    const float* efeat = (const float*)d_in[0];
    const float* gfeat = (const float*)d_in[1];
    const float* mfeat = (const float*)d_in[2];
    const int*   src   = (const int*)d_in[3];
    const int*   dst   = (const int*)d_in[4];
    const float* ew1 = (const float*)d_in[5];
    const float* eb1 = (const float*)d_in[6];
    const float* ew2 = (const float*)d_in[7];
    const float* eb2 = (const float*)d_in[8];
    const float* eg  = (const float*)d_in[9];
    const float* ebv = (const float*)d_in[10];
    const float* sw1 = (const float*)d_in[11];
    const float* sb1 = (const float*)d_in[12];
    const float* sw2 = (const float*)d_in[13];
    const float* sb2 = (const float*)d_in[14];
    const float* sg  = (const float*)d_in[15];
    const float* sbv = (const float*)d_in[16];
    const float* dw1 = (const float*)d_in[17];
    const float* db1 = (const float*)d_in[18];
    const float* dw2 = (const float*)d_in[19];
    const float* db2 = (const float*)d_in[20];
    const float* dg  = (const float*)d_in[21];
    const float* dbv = (const float*)d_in[22];

    // workspace: f16 transposed weights ([N][K] layout), ~288 KB total
    half_t* ws   = (half_t*)d_ws;
    half_t* ew1t = ws;               // 384*128 = 49152
    half_t* ew2t = ws + 49152;       // 16384
    half_t* sw1t = ws + 65536;       // 16384
    half_t* sw2t = ws + 81920;       // 16384
    half_t* dw1t = ws + 98304;       // 256*128 = 32768
    half_t* dw2t = ws + 131072;      // 16384

    float* grid_out = (float*)d_out;
    float* mesh_out = (float*)d_out + (size_t)NGRID * DIM;  // doubles as agg

    transpose_f16<<<(49152 + 255) / 256, 256, 0, stream>>>(ew1, ew1t, 384, 128, 49152);
    transpose_f16<<<(16384 + 255) / 256, 256, 0, stream>>>(ew2, ew2t, 128, 128, 16384);
    transpose_f16<<<(16384 + 255) / 256, 256, 0, stream>>>(sw1, sw1t, 128, 128, 16384);
    transpose_f16<<<(16384 + 255) / 256, 256, 0, stream>>>(sw2, sw2t, 128, 128, 16384);
    transpose_f16<<<(32768 + 255) / 256, 256, 0, stream>>>(dw1, dw1t, 256, 128, 32768);
    transpose_f16<<<(16384 + 255) / 256, 256, 0, stream>>>(dw2, dw2t, 128, 128, 16384);

    zero_f32<<<(NMESH * DIM + 255) / 256, 256, 0, stream>>>(mesh_out, NMESH * DIM);

    edge_mlp_scatter<<<E_TOT / 32, 256, 0, stream>>>(
        efeat, gfeat, mfeat, src, dst,
        ew1t, eb1, ew2t, eb2, eg, ebv, mesh_out);

    grid_mlp<<<NGRID / 32, 256, 0, stream>>>(
        gfeat, sw1t, sb1, sw2t, sb2, sg, sbv, grid_out);

    mesh_mlp<<<NMESH / 32, 256, 0, stream>>>(
        mesh_out, mfeat, dw1t, db1, dw2t, db2, dg, dbv, mesh_out);
}
